// ProgrammableMatterGNN_86088324481462
// MI455X (gfx1250) — matrix operand order repaired and checked by reading, not compiled, not run on hardware
//
#include <hip/hip_runtime.h>

typedef __attribute__((ext_vector_type(16))) __bf16 v16bf;
typedef __attribute__((ext_vector_type(8)))  float  v8f;

#define GCN_HID 64
#define NGRAPH  256

template <bool RELU>
__device__ __forceinline__ __bf16 cvt_bf16(float x) {
  if (RELU) x = fmaxf(x, 0.0f);
  return (__bf16)x;
}

// ---------------------------------------------------------------------------
// WMMA GEMM, fully specialized: out[M,N] = (relu?)(A)[M,K] x W[K,N] (+bias)
// K in {8,32,64}, N in {1,9,32,64}. One wave32 per 16-row tile, NT col tiles.
// A fragment = two contiguous 8-float runs per lane -> float4 loads.
// All guards are compile-time or branch-free value selects (EXEC stays full
// around the WMMA as the ISA requires).
// ---------------------------------------------------------------------------
template <int K, int N, bool RELU_IN, bool RELU_OUT, bool BIAS>
__global__ void wmma_gemm_kernel(const float* __restrict__ A,
                                 const float* __restrict__ W,
                                 const float* __restrict__ bias,
                                 float* __restrict__ out, int M) {
  constexpr int NT = (N + 15) / 16;
  constexpr int KTILES = (K + 31) / 32;
  const int wave = threadIdx.x >> 5;
  const int lane = threadIdx.x & 31;
  const int tile = blockIdx.x * 4 + wave;        // 16-row tile
  if (tile * 16 >= M) return;                    // wave-uniform exit
  const int kSel = lane >> 4;                    // lane half
  const int l15  = lane & 15;
  const int mRow = tile * 16 + l15;
  const float* __restrict__ arow = A + (size_t)mRow * K;

  v8f zero = {0.f, 0.f, 0.f, 0.f, 0.f, 0.f, 0.f, 0.f};
  v8f acc[NT];
#pragma unroll
  for (int t = 0; t < NT; ++t) acc[t] = zero;

#pragma unroll
  for (int kt = 0; kt < KTILES * 32; kt += 32) {
    // ---- A fragment (16x32 bf16): elements 0..7 = k kt+kSel*8..+7,
    // ----                          elements 8..15 = k kt+16+kSel*8..+7
    v16bf a;
    if constexpr (K >= 32) {
      float4 q0 = *(const float4*)(arow + kt + kSel * 8);
      float4 q1 = *(const float4*)(arow + kt + kSel * 8 + 4);
      float4 q2 = *(const float4*)(arow + kt + 16 + kSel * 8);
      float4 q3 = *(const float4*)(arow + kt + 16 + kSel * 8 + 4);
      a[0]  = cvt_bf16<RELU_IN>(q0.x); a[1]  = cvt_bf16<RELU_IN>(q0.y);
      a[2]  = cvt_bf16<RELU_IN>(q0.z); a[3]  = cvt_bf16<RELU_IN>(q0.w);
      a[4]  = cvt_bf16<RELU_IN>(q1.x); a[5]  = cvt_bf16<RELU_IN>(q1.y);
      a[6]  = cvt_bf16<RELU_IN>(q1.z); a[7]  = cvt_bf16<RELU_IN>(q1.w);
      a[8]  = cvt_bf16<RELU_IN>(q2.x); a[9]  = cvt_bf16<RELU_IN>(q2.y);
      a[10] = cvt_bf16<RELU_IN>(q2.z); a[11] = cvt_bf16<RELU_IN>(q2.w);
      a[12] = cvt_bf16<RELU_IN>(q3.x); a[13] = cvt_bf16<RELU_IN>(q3.y);
      a[14] = cvt_bf16<RELU_IN>(q3.z); a[15] = cvt_bf16<RELU_IN>(q3.w);
    } else {
      // K == 8: only kSel==0 lanes, elements 0..7 carry data (k=0..7).
      float4 q0 = *(const float4*)(arow);
      float4 q1 = *(const float4*)(arow + 4);
      const bool ok = (kSel == 0);
      a[0] = ok ? cvt_bf16<RELU_IN>(q0.x) : (__bf16)0.f;
      a[1] = ok ? cvt_bf16<RELU_IN>(q0.y) : (__bf16)0.f;
      a[2] = ok ? cvt_bf16<RELU_IN>(q0.z) : (__bf16)0.f;
      a[3] = ok ? cvt_bf16<RELU_IN>(q0.w) : (__bf16)0.f;
      a[4] = ok ? cvt_bf16<RELU_IN>(q1.x) : (__bf16)0.f;
      a[5] = ok ? cvt_bf16<RELU_IN>(q1.y) : (__bf16)0.f;
      a[6] = ok ? cvt_bf16<RELU_IN>(q1.z) : (__bf16)0.f;
      a[7] = ok ? cvt_bf16<RELU_IN>(q1.w) : (__bf16)0.f;
#pragma unroll
      for (int i = 8; i < 16; ++i) a[i] = (__bf16)0.f;
    }

#pragma unroll
    for (int t = 0; t < NT; ++t) {
      // ---- B fragment (32x16 bf16): same k order as A,
      // ---- element 2v,2v+1 = W[kt+kSel*8+2(v&3)+16(v>>2)..+1][nCol]
      const int nCol = t * 16 + l15;
      const bool nok = (N % 16 == 0) || (nCol < N);
      const int  ns  = nok ? nCol : 0;   // clamped (always-valid) column
      v16bf b;
#pragma unroll
      for (int v = 0; v < 8; ++v) {
        const int k0 = kt + kSel * 8 + 2 * (v & 3) + (v >> 2) * 16;
        const bool kok0 = (K % 32 == 0) || (k0 < K);
        const bool kok1 = (K % 32 == 0) || (k0 + 1 < K);
        const int  ks0  = kok0 ? k0 : 0;
        const int  ks1  = kok1 ? (k0 + 1) : 0;
        float w0 = W[ks0 * N + ns];
        float w1 = W[ks1 * N + ns];
        b[2 * v]     = (kok0 && nok) ? (__bf16)w0 : (__bf16)0.f;
        b[2 * v + 1] = (kok1 && nok) ? (__bf16)w1 : (__bf16)0.f;
      }
      acc[t] = __builtin_amdgcn_wmma_f32_16x16x32_bf16(
          false, a, false, b, (short)0, acc[t], false, false);
    }
  }

  // ---- D store: lane half picks M group (v + 8*kSel), l15 = column
#pragma unroll
  for (int t = 0; t < NT; ++t) {
    const int nCol = t * 16 + l15;
    if ((N % 16 == 0) || (nCol < N)) {
      float bv = 0.0f;
      if constexpr (BIAS) bv = bias[nCol];
#pragma unroll
      for (int v = 0; v < 8; ++v) {
        const int row = tile * 16 + v + 8 * kSel;
        float r = acc[t][v];
        if constexpr (BIAS) r += bv;          // no +0.0f VALU for non-bias GEMMs
        if constexpr (RELU_OUT) r = fmaxf(r, 0.0f);
        out[(size_t)row * N + nCol] = r;
      }
    }
  }
}

// ---------------------------------------------------------------------------
// Degree / normalization
// ---------------------------------------------------------------------------
__global__ void deg_init_kernel(float* deg, int n) {
  int i = blockIdx.x * blockDim.x + threadIdx.x;
  if (i < n) deg[i] = 1.0f;  // self loop
}

__global__ void deg_edge_kernel(const int* __restrict__ dst, float* deg, int E) {
  int e = blockIdx.x * blockDim.x + threadIdx.x;
  if (e < E) unsafeAtomicAdd(&deg[dst[e]], 1.0f);
}

__global__ void dinv_kernel(float* deg, int n) {
  int i = blockIdx.x * blockDim.x + threadIdx.x;
  if (i < n) deg[i] = rsqrtf(deg[i]);  // deg >= 1 always
}

// agg[i, q..q+3] = h[i, q..q+3]*dinv[i]^2 + b[q..q+3]  (self loop + bias)
__global__ void aggr_init_kernel(const float* __restrict__ h,
                                 const float* __restrict__ dinv,
                                 const float* __restrict__ bias,
                                 float* __restrict__ agg, int n) {
  int idx = blockIdx.x * blockDim.x + threadIdx.x;   // n*16 threads
  if (idx >= n * 16) return;
  int i = idx >> 4;
  int q = (idx & 15) * 4;
  float di = dinv[i];
  float w = di * di;
  float4 hv = *(const float4*)(h + (size_t)i * GCN_HID + q);
  float4 bv = *(const float4*)(bias + q);
  float4 r;
  r.x = hv.x * w + bv.x;
  r.y = hv.y * w + bv.y;
  r.z = hv.z * w + bv.z;
  r.w = hv.w * w + bv.w;
  *(float4*)(agg + (size_t)i * GCN_HID + q) = r;
}

// agg[d, q..q+3] += h[s, q..q+3] * dinv[s]*dinv[d]  (16 threads per edge)
__global__ void edge_aggr_kernel(const int* __restrict__ src,
                                 const int* __restrict__ dst,
                                 const float* __restrict__ h,
                                 const float* __restrict__ dinv,
                                 float* __restrict__ agg, int E) {
  long long idx = (long long)blockIdx.x * blockDim.x + threadIdx.x;
  int e = (int)(idx >> 4);
  int q = (int)(idx & 15) * 4;
  if (e >= E) return;
  int s = src[e];
  int d = dst[e];
  float w = dinv[s] * dinv[d];
  float4 hv = *(const float4*)(h + (size_t)s * GCN_HID + q);
  float* p = agg + (size_t)d * GCN_HID + q;
  unsafeAtomicAdd(p + 0, hv.x * w);
  unsafeAtomicAdd(p + 1, hv.y * w);
  unsafeAtomicAdd(p + 2, hv.z * w);
  unsafeAtomicAdd(p + 3, hv.w * w);
}

// ---------------------------------------------------------------------------
// Mean pooling over graphs
// ---------------------------------------------------------------------------
__global__ void zero_kernel(float* p, int n) {
  int i = blockIdx.x * blockDim.x + threadIdx.x;
  if (i < n) p[i] = 0.0f;
}

__global__ void pool_accum_kernel(const int* __restrict__ batch,
                                  const float* __restrict__ h,
                                  float* __restrict__ sums,
                                  float* __restrict__ cnts, int n) {
  int idx = blockIdx.x * blockDim.x + threadIdx.x;   // n*16 threads
  if (idx >= n * 16) return;
  int i = idx >> 4;
  int q = (idx & 15) * 4;
  int g = batch[i];
  float4 hv = *(const float4*)(h + (size_t)i * GCN_HID + q);
  float* p = sums + (size_t)g * GCN_HID + q;
  unsafeAtomicAdd(p + 0, hv.x);
  unsafeAtomicAdd(p + 1, hv.y);
  unsafeAtomicAdd(p + 2, hv.z);
  unsafeAtomicAdd(p + 3, hv.w);
  if (q == 0) unsafeAtomicAdd(&cnts[g], 1.0f);
}

__global__ void pool_div_kernel(const float* __restrict__ sums,
                                const float* __restrict__ cnts,
                                float* __restrict__ xg, int total) {
  int idx = blockIdx.x * blockDim.x + threadIdx.x;
  if (idx >= total) return;
  int g = idx >> 6;
  xg[idx] = sums[idx] / fmaxf(cnts[g], 1.0f);
}

// ---------------------------------------------------------------------------
extern "C" void kernel_launch(void* const* d_in, const int* in_sizes, int n_in,
                              void* d_out, int out_size, void* d_ws, size_t ws_size,
                              hipStream_t stream) {
  const float* x     = (const float*)d_in[0];
  const int*   ei    = (const int*)d_in[1];
  const int*   batch = (const int*)d_in[2];
  const float* W1 = (const float*)d_in[3],  *b1  = (const float*)d_in[4];
  const float* W2 = (const float*)d_in[5],  *b2  = (const float*)d_in[6];
  const float* W3 = (const float*)d_in[7],  *b3  = (const float*)d_in[8];
  const float* Wg = (const float*)d_in[9],  *bg  = (const float*)d_in[10];
  const float* Wh1 = (const float*)d_in[11], *bh1 = (const float*)d_in[12];
  const float* Wh2 = (const float*)d_in[13], *bh2 = (const float*)d_in[14];
  const float* Wm1 = (const float*)d_in[15], *bm1 = (const float*)d_in[16];
  const float* Wm2 = (const float*)d_in[17], *bm2 = (const float*)d_in[18];
  const float* Wp1 = (const float*)d_in[19], *bp1 = (const float*)d_in[20];
  const float* Wp2 = (const float*)d_in[21], *bp2 = (const float*)d_in[22];

  const int N = in_sizes[0] / 8;     // 100000 nodes (multiple of 16)
  const int E = in_sizes[1] / 2;     // 3.2M edges
  const int* src = ei;
  const int* dst = ei + E;

  // Workspace layout (floats)
  float* ws = (float*)d_ws;
  size_t off = 0;
  float* dinv = ws + off; off += (size_t)N;
  float* B0   = ws + off; off += (size_t)N * GCN_HID;   // GEMM output h
  float* B1   = ws + off; off += (size_t)N * GCN_HID;   // aggregation ping
  float* B2   = ws + off; off += (size_t)N * GCN_HID;   // aggregation pong
  float* sums = ws + off; off += (size_t)NGRAPH * GCN_HID;
  float* cnts = ws + off; off += (size_t)NGRAPH;
  float* xgin = ws + off; off += (size_t)NGRAPH * GCN_HID;
  float* xg   = ws + off; off += (size_t)NGRAPH * GCN_HID;
  float* t1   = ws + off; off += (size_t)NGRAPH * (GCN_HID / 2);

  float* heur  = (float*)d_out;                           // [256,1]
  float* mlog  = (float*)d_out + NGRAPH;                  // [N,9]
  float* bprio = (float*)d_out + NGRAPH + (size_t)N * 9;  // [N,1]

  const int T = 256;
  const int nqBlocks   = (N * 16 + T - 1) / T;
  const int eqBlocks   = (int)(((long long)E * 16 + T - 1) / T);
  const int nodeTiles  = (N + 15) / 16;
  const int gemmBlocks = (nodeTiles + 3) / 4;             // 4 waves / block
  const int headBlocks = ((NGRAPH / 16) + 3) / 4;

  // --- normalization: deg = indeg + 1, dinv = rsqrt(deg) ---
  deg_init_kernel<<<(N + T - 1) / T, T, 0, stream>>>(dinv, N);
  deg_edge_kernel<<<(E + T - 1) / T, T, 0, stream>>>(dst, dinv, E);
  dinv_kernel<<<(N + T - 1) / T, T, 0, stream>>>(dinv, N);

  // --- layer 1: h = x@W1 ; agg1 = D^-1/2 (A+I) D^-1/2 h + b1 ---
  wmma_gemm_kernel<8, 64, false, false, false><<<gemmBlocks, 128, 0, stream>>>(x, W1, nullptr, B0, N);
  aggr_init_kernel<<<nqBlocks, T, 0, stream>>>(B0, dinv, b1, B1, N);
  edge_aggr_kernel<<<eqBlocks, T, 0, stream>>>(src, dst, B0, dinv, B1, E);

  // --- layer 2 (relu folded into A-load) ---
  wmma_gemm_kernel<64, 64, true, false, false><<<gemmBlocks, 128, 0, stream>>>(B1, W2, nullptr, B0, N);
  aggr_init_kernel<<<nqBlocks, T, 0, stream>>>(B0, dinv, b2, B2, N);
  edge_aggr_kernel<<<eqBlocks, T, 0, stream>>>(src, dst, B0, dinv, B2, E);

  // --- layer 3 -> h3 in B1 ---
  wmma_gemm_kernel<64, 64, true, false, false><<<gemmBlocks, 128, 0, stream>>>(B2, W3, nullptr, B0, N);
  aggr_init_kernel<<<nqBlocks, T, 0, stream>>>(B0, dinv, b3, B1, N);
  edge_aggr_kernel<<<eqBlocks, T, 0, stream>>>(src, dst, B0, dinv, B1, E);

  // --- global mean pool ---
  zero_kernel<<<(NGRAPH * GCN_HID + NGRAPH + T - 1) / T, T, 0, stream>>>(sums, NGRAPH * GCN_HID + NGRAPH);
  pool_accum_kernel<<<nqBlocks, T, 0, stream>>>(batch, B1, sums, cnts, N);
  pool_div_kernel<<<(NGRAPH * GCN_HID + T - 1) / T, T, 0, stream>>>(sums, cnts, xgin, NGRAPH * GCN_HID);

  // --- heuristic head ---
  wmma_gemm_kernel<64, 64, false, true,  true><<<headBlocks, 128, 0, stream>>>(xgin, Wg,  bg,  xg,   NGRAPH);
  wmma_gemm_kernel<64, 32, false, true,  true><<<headBlocks, 128, 0, stream>>>(xg,   Wh1, bh1, t1,   NGRAPH);
  wmma_gemm_kernel<32, 1,  false, false, true><<<headBlocks, 128, 0, stream>>>(t1,   Wh2, bh2, heur, NGRAPH);

  // --- move_logits head ---
  wmma_gemm_kernel<64, 64, false, true,  true><<<gemmBlocks, 128, 0, stream>>>(B1, Wm1, bm1, B0,   N);
  wmma_gemm_kernel<64, 9,  false, false, true><<<gemmBlocks, 128, 0, stream>>>(B0, Wm2, bm2, mlog, N);

  // --- block_priority head ---
  wmma_gemm_kernel<64, 32, false, true,  true><<<gemmBlocks, 128, 0, stream>>>(B1, Wp1, bp1, B2,    N);
  wmma_gemm_kernel<32, 1,  false, false, true><<<gemmBlocks, 128, 0, stream>>>(B2, Wp2, bp2, bprio, N);
}
